// Ex2_73753178407685
// MI455X (gfx1250) — matrix operand order repaired and checked by reading, not compiled, not run on hardware
//
#include <hip/hip_runtime.h>
#include <math.h>

// ---------------- problem constants ----------------
#define BATCH   2
#define SEQ     9000
#define TLEN    8192
#define CROPOFF 404        // (SEQ - TLEN) / 2
#define FEAT_CH 224        // 128 (enc1) + 64 (enc2) + 32 (feat2)
#define NBUCKET 128        // TLEN / 64

typedef __attribute__((ext_vector_type(16))) _Float16 v16h;
typedef __attribute__((ext_vector_type(4)))  _Float16 v4h;
typedef __attribute__((ext_vector_type(8)))  float    v8f;

union Frag16 { v16h v; unsigned int u[8]; };

enum { ACT_NONE = 0, ACT_RELU = 1, ACT_GELU = 2 };

// ---------------------------------------------------------------------
// Optional CDNA5 async global->LDS copies (ASYNCcnt path), compile-safe.
// The clang builtins take pointers in the CUDA __device__/__shared__
// language address spaces, which cannot be spelled in HIP source; we
// therefore declare the underlying LLVM intrinsics directly (opaque
// pointers: only the address spaces 1 / 3 must match).  AS(3) pointer
// value = low 32 bits of the generic LDS address (aperture rule).
// ---------------------------------------------------------------------
#if __has_builtin(__builtin_amdgcn_global_load_async_to_lds_b128) && \
    __has_builtin(__builtin_amdgcn_global_load_async_to_lds_b32)  && \
    __has_builtin(__builtin_amdgcn_s_wait_asynccnt)
#define HAVE_ASYNC_LDS 1
typedef __attribute__((address_space(1))) char as1_char;
typedef __attribute__((address_space(3))) char as3_char;
extern "C" __device__ void
amdgcn_async_load_b128(as1_char* gsrc, as3_char* lds, int off, int cpol)
    __asm("llvm.amdgcn.global.load.async.to.lds.b128");
extern "C" __device__ void
amdgcn_async_load_b32(as1_char* gsrc, as3_char* lds, int off, int cpol)
    __asm("llvm.amdgcn.global.load.async.to.lds.b32");
__device__ __forceinline__ void async_copy_b128(void* lds_dst, const void* gsrc) {
  amdgcn_async_load_b128((as1_char*)(unsigned long long)gsrc,
                         (as3_char*)(unsigned)(unsigned long long)lds_dst, 0, 0);
}
__device__ __forceinline__ void async_copy_b32(void* lds_dst, const void* gsrc) {
  amdgcn_async_load_b32((as1_char*)(unsigned long long)gsrc,
                        (as3_char*)(unsigned)(unsigned long long)lds_dst, 0, 0);
}
__device__ __forceinline__ void async_wait() {
  __builtin_amdgcn_s_wait_asynccnt(0);
}
#else
#define HAVE_ASYNC_LDS 0
__device__ __forceinline__ void async_copy_b128(void* lds_dst, const void* gsrc) {
  *reinterpret_cast<float4*>(lds_dst) = *reinterpret_cast<const float4*>(gsrc);
}
__device__ __forceinline__ void async_copy_b32(void* lds_dst, const void* gsrc) {
  *reinterpret_cast<float*>(lds_dst) = *reinterpret_cast<const float*>(gsrc);
}
__device__ __forceinline__ void async_wait() {}
#endif

__device__ __forceinline__ float act_apply(float v, int act) {
  if (act == ACT_RELU) return fmaxf(v, 0.f);
  if (act == ACT_GELU) return 0.5f * v * (1.f + erff(v * 0.70710678118654752f));
  return v;
}

// ---------------------------------------------------------------------
// LDS staging for a 64(M) x 32(K) A-tile and 32(K) x 64(N) B-tile.
// Fast path (block-uniform full tile): float4 global loads, vector LDS
// stores, no per-lane control flow. Slow path: clamped addresses +
// select (v_cndmask), still unconditional loads.
// As layout: [m][k] (row = 32 halfs).  Bs layout: [n][k] (transposed).
// ---------------------------------------------------------------------
__device__ __forceinline__ void gemm_stage(
    _Float16* __restrict__ As, _Float16* __restrict__ Bs,
    const float* __restrict__ Wm, const float* __restrict__ Xb,
    int M, int K, int N, int ldx, int m0, int n0, int k0, int tid)
{
  const bool fullA = (m0 + 64 <= M) && (k0 + 32 <= K);
  const bool fullB = (n0 + 64 <= N) && (k0 + 32 <= K);
  if (fullA) {
    #pragma unroll
    for (int i = 0; i < 2; ++i) {
      int g = tid + (i << 8);
      int mm = g >> 3, kk = (g & 7) << 2;
      const float4 v =
          *reinterpret_cast<const float4*>(Wm + (long)(m0 + mm) * K + (k0 + kk));
      v4h h = {(_Float16)v.x, (_Float16)v.y, (_Float16)v.z, (_Float16)v.w};
      *reinterpret_cast<v4h*>(As + mm * 32 + kk) = h;
    }
  } else {
    #pragma unroll
    for (int i = 0; i < 8; ++i) {
      int g = tid + (i << 8);
      int mm = g >> 5, kk = g & 31;
      int gm = m0 + mm, gk = k0 + kk;
      int cm = gm < M ? gm : M - 1;
      int ck = gk < K ? gk : K - 1;
      float v = Wm[(long)cm * K + ck];
      As[mm * 32 + kk] = (_Float16)((gm < M && gk < K) ? v : 0.f);
    }
  }
  if (fullB) {
    #pragma unroll
    for (int i = 0; i < 2; ++i) {
      int g = tid + (i << 8);
      int kk = g >> 4, nn = (g & 15) << 2;
      const float4 v = *reinterpret_cast<const float4*>(
          Xb + (long)(k0 + kk) * ldx + (n0 + nn));
      Bs[(nn + 0) * 32 + kk] = (_Float16)v.x;
      Bs[(nn + 1) * 32 + kk] = (_Float16)v.y;
      Bs[(nn + 2) * 32 + kk] = (_Float16)v.z;
      Bs[(nn + 3) * 32 + kk] = (_Float16)v.w;
    }
  } else {
    #pragma unroll
    for (int i = 0; i < 8; ++i) {
      int g = tid + (i << 8);
      int nn = g & 63, kk = g >> 6;
      int gn = n0 + nn, gk = k0 + kk;
      int cn = gn < N ? gn : N - 1;
      int ck = gk < K ? gk : K - 1;
      float v = Xb[(long)ck * ldx + cn];
      Bs[nn * 32 + kk] = (_Float16)((gn < N && gk < K) ? v : 0.f);
    }
  }
}

// =====================================================================
// Generic GEMM: Y[M x N] = act(W[M x K] * X[K x N] + bias) (+= old Y)
// 64x64 tile per 256-thread block; 8 wave32 waves as 4(m) x 2(n), each
// wave holds two 16x16 f32 accumulators (A fragment reused).  LDS is
// double-buffered: one barrier per K-chunk, loads overlap WMMA.
// =====================================================================
__global__ __launch_bounds__(256) void wmma_gemm_kernel(
    const float* __restrict__ Wm,   // M x K row-major (shared over batch)
    const float* __restrict__ X,    // per batch: K x N, row stride ldx
    const float* __restrict__ bias, // M or nullptr
    float* __restrict__ Y,          // per batch: M x N, row stride ldy
    int M, int K, int N, int ldx, int ldy,
    long sxb, long syb, int act, int accum)
{
  __shared__ _Float16 As[2][64 * 32];
  __shared__ _Float16 Bs[2][64 * 32];
  const float* Xb = X + (long)blockIdx.z * sxb;
  float* Yb = Y + (long)blockIdx.z * syb;
  const int m0 = blockIdx.y * 64;
  const int n0 = blockIdx.x * 64;
  const int tid  = threadIdx.x;
  const int lane = tid & 31;
  const int wave = tid >> 5;
  const int wm = wave >> 1;          // 0..3  (M sub-tile)
  const int wn = wave & 1;           // 0..1  (pair of N sub-tiles)
  const int hi = lane >> 4;
  v8f acc0 = {}, acc1 = {};
  const int nK = (K + 31) >> 5;
  gemm_stage(As[0], Bs[0], Wm, Xb, M, K, N, ldx, m0, n0, 0, tid);
  for (int kc = 0; kc < nK; ++kc) {
    __syncthreads();
    const int cur = kc & 1;
    if (kc + 1 < nK)
      gemm_stage(As[cur ^ 1], Bs[cur ^ 1], Wm, Xb, M, K, N, ldx,
                 m0, n0, (kc + 1) << 5, tid);
    if (kc + 2 < nK) {   // warm the chunk after next (global_prefetch_b8)
      __builtin_prefetch(Wm + (long)m0 * K + ((kc + 2) << 5), 0, 1);
      __builtin_prefetch(Xb + (long)((kc + 2) << 5) * ldx + n0, 0, 1);
    }
    Frag16 fa, fb0, fb1;
    const unsigned int* A32 = (const unsigned int*)As[cur];
    const unsigned int* B32 = (const unsigned int*)Bs[cur];
    const int mrow = wm * 16 + (lane & 15);
    #pragma unroll
    for (int p = 0; p < 8; ++p) {   // A frag dword order (K pairs)
      int dk = (hi << 3) + p;
      fa.u[p] = A32[mrow * 16 + dk];
    }
    const int nrow0 = wn * 32 + (lane & 15);
    #pragma unroll
    for (int p = 0; p < 8; ++p) {   // B frags: contiguous dwords per lane
      fb0.u[p] = B32[nrow0 * 16 + hi * 8 + p];
      fb1.u[p] = B32[(nrow0 + 16) * 16 + hi * 8 + p];
    }
    acc0 = __builtin_amdgcn_wmma_f32_16x16x32_f16(false, fa.v, false, fb0.v,
                                                  (short)0, acc0, false, false);
    acc1 = __builtin_amdgcn_wmma_f32_16x16x32_f16(false, fa.v, false, fb1.v,
                                                  (short)0, acc1, false, false);
  }
  const int gn0 = n0 + wn * 32 + (lane & 15);
  const int gn1 = gn0 + 16;
  const int mbase = m0 + wm * 16 + hi * 8;
  #pragma unroll
  for (int r = 0; r < 8; ++r) {
    int gm = mbase + r;
    if (gm < M) {
      float bsv = bias ? bias[gm] : 0.f;
      if (gn0 < N) {
        float v = act_apply(acc0[r] + bsv, act);
        long o = (long)gm * ldy + gn0;
        if (accum) v += Yb[o];
        Yb[o] = v;
      }
      if (gn1 < N) {
        float v = act_apply(acc1[r] + bsv, act);
        long o = (long)gm * ldy + gn1;
        if (accum) v += Yb[o];
        Yb[o] = v;
      }
    }
  }
}

// ---------------------------------------------------------------------
// Dilated-conv staging: A = W[o, c, tap] (stride `taps`, weights are
// L2-resident), B = X columns shifted by (tap-(T-1)/2)*ar.  The shift
// window test is block-uniform, so edge handling costs no divergence.
// ---------------------------------------------------------------------
__device__ __forceinline__ void dconv_stage(
    _Float16* __restrict__ As, _Float16* __restrict__ Bs,
    const float* __restrict__ Wt, const float* __restrict__ Xb,
    int C, int N, int taps, int m0, int n0, int k0, int tap, int shift, int tid)
{
  #pragma unroll
  for (int i = 0; i < 2; ++i) {
    int g = tid + (i << 8);
    int mm = g >> 3, kk = (g & 7) << 2;
    const float* wp = Wt + ((long)(m0 + mm) * C + (k0 + kk)) * taps + tap;
    v4h h = {(_Float16)wp[0], (_Float16)wp[(long)taps],
             (_Float16)wp[2L * taps], (_Float16)wp[3L * taps]};
    *reinterpret_cast<v4h*>(As + mm * 32 + kk) = h;
  }
  const int s0 = n0 + shift;
  if (s0 >= 0 && s0 + 64 <= N) {
    #pragma unroll
    for (int i = 0; i < 8; ++i) {
      int g = tid + (i << 8);
      int nn = g & 63, kk = g >> 6;
      Bs[nn * 32 + kk] = (_Float16)Xb[(long)(k0 + kk) * N + s0 + nn];
    }
  } else {
    #pragma unroll
    for (int i = 0; i < 8; ++i) {
      int g = tid + (i << 8);
      int nn = g & 63, kk = g >> 6;
      int src = s0 + nn;
      int cs = src < 0 ? 0 : (src < N ? src : N - 1);
      float v = Xb[(long)(k0 + kk) * N + cs];
      Bs[nn * 32 + kk] = (_Float16)((src >= 0 && src < N) ? v : 0.f);
    }
  }
}

// =====================================================================
// Dilated conv as per-tap implicit GEMM, double-buffered over the
// linearized (tap x K-chunk) pipeline.
// =====================================================================
__global__ __launch_bounds__(256) void wmma_dconv_kernel(
    const float* __restrict__ Wt,   // C x C x taps
    const float* __restrict__ X,    // per batch: C x N
    const float* __restrict__ bias, // C
    float* __restrict__ Y,          // per batch: C x N
    int C, int N, int taps, int ar, int accum)
{
  __shared__ _Float16 As[2][64 * 32];
  __shared__ _Float16 Bs[2][64 * 32];
  const long cs = (long)C * N;
  const float* Xb = X + (long)blockIdx.z * cs;
  float* Yb = Y + (long)blockIdx.z * cs;
  const int m0 = blockIdx.y * 64;
  const int n0 = blockIdx.x * 64;
  const int tid  = threadIdx.x;
  const int lane = tid & 31;
  const int wave = tid >> 5;
  const int wm = wave >> 1, wn = wave & 1;
  const int hi = lane >> 4;
  const int half = (taps - 1) >> 1;
  v8f acc0 = {}, acc1 = {};
  const int nK = C >> 5;             // C is 64 or 128
  const int nQ = taps * nK;
  dconv_stage(As[0], Bs[0], Wt, Xb, C, N, taps, m0, n0, 0, 0, -half * ar, tid);
  for (int q = 0; q < nQ; ++q) {
    __syncthreads();
    const int cur = q & 1;
    if (q + 1 < nQ) {
      int t2 = (q + 1) / nK, kc2 = (q + 1) % nK;
      dconv_stage(As[cur ^ 1], Bs[cur ^ 1], Wt, Xb, C, N, taps,
                  m0, n0, kc2 << 5, t2, (t2 - half) * ar, tid);
    }
    Frag16 fa, fb0, fb1;
    const unsigned int* A32 = (const unsigned int*)As[cur];
    const unsigned int* B32 = (const unsigned int*)Bs[cur];
    const int mrow = wm * 16 + (lane & 15);
    #pragma unroll
    for (int p = 0; p < 8; ++p) {
      int dk = (hi << 3) + p;
      fa.u[p] = A32[mrow * 16 + dk];
    }
    const int nrow0 = wn * 32 + (lane & 15);
    #pragma unroll
    for (int p = 0; p < 8; ++p) {
      fb0.u[p] = B32[nrow0 * 16 + hi * 8 + p];
      fb1.u[p] = B32[(nrow0 + 16) * 16 + hi * 8 + p];
    }
    acc0 = __builtin_amdgcn_wmma_f32_16x16x32_f16(false, fa.v, false, fb0.v,
                                                  (short)0, acc0, false, false);
    acc1 = __builtin_amdgcn_wmma_f32_16x16x32_f16(false, fa.v, false, fb1.v,
                                                  (short)0, acc1, false, false);
  }
  const int gn0 = n0 + wn * 32 + (lane & 15);
  const int gn1 = gn0 + 16;
  const int mbase = m0 + wm * 16 + hi * 8;
  #pragma unroll
  for (int r = 0; r < 8; ++r) {
    int gm = mbase + r;
    float bsv = bias[gm];
    if (gn0 < N) {
      float v = acc0[r] + bsv;
      long o = (long)gm * N + gn0;
      if (accum) v += Yb[o];
      Yb[o] = v;
    }
    if (gn1 < N) {
      float v = acc1[r] + bsv;
      long o = (long)gm * N + gn1;
      if (accum) v += Yb[o];
      Yb[o] = v;
    }
  }
}

// =====================================================================
// Elementwise / small kernels (float4 vectorized where strides allow)
// =====================================================================
__global__ __launch_bounds__(256) void bnrelu_kernel(
    const float* __restrict__ x, const float* __restrict__ g,
    const float* __restrict__ be, const float* __restrict__ rm,
    const float* __restrict__ rv, float* __restrict__ y, int L)
{
  long i4 = (long)blockIdx.x * 256 + threadIdx.x;
  long total4 = (long)BATCH * L * (SEQ / 4);
  if (i4 >= total4) return;
  long base = i4 * 4;
  int c = (int)((base / SEQ) % L);     // SEQ % 4 == 0 -> uniform over the 4
  float s = g[c] * rsqrtf(rv[c] + 1e-5f);
  float sh = be[c] - rm[c] * s;
  float4 v = reinterpret_cast<const float4*>(x)[i4];
  float4 o;
  o.x = fmaxf(v.x * s + sh, 0.f);
  o.y = fmaxf(v.y * s + sh, 0.f);
  o.z = fmaxf(v.z * s + sh, 0.f);
  o.w = fmaxf(v.w * s + sh, 0.f);
  reinterpret_cast<float4*>(y)[i4] = o;
}

__global__ __launch_bounds__(256) void copy_crop_kernel(
    const float* __restrict__ src, float* __restrict__ feat,
    int Lsrc, int nch, int chOff)
{
  long i4 = (long)blockIdx.x * 256 + threadIdx.x;
  long total4 = (long)BATCH * nch * (TLEN / 4);
  if (i4 >= total4) return;
  long base = i4 * 4;
  int t = (int)(base % TLEN);
  long r = base / TLEN;
  int c = (int)(r % nch);
  int b = (int)(r / nch);
  const float4 v = *reinterpret_cast<const float4*>(
      src + ((long)b * Lsrc + c) * SEQ + t + CROPOFF);   // CROPOFF % 4 == 0
  *reinterpret_cast<float4*>(
      feat + ((long)b * FEAT_CH + chOff + c) * TLEN + t) = v;
}

__global__ __launch_bounds__(256) void add_pe_kernel(
    float* __restrict__ H, const float* __restrict__ pr,
    const float* __restrict__ pc)
{
  long i4 = (long)blockIdx.x * 256 + threadIdx.x;
  if (i4 >= (long)BATCH * 64 * (TLEN / 4)) return;
  long base = i4 * 4;
  int t = (int)(base % TLEN);
  int c = (int)((base / TLEN) % 64);
  float rowv = pr[(t >> 6) * 64 + c];        // t>>6 uniform over the 4
  float4 v = reinterpret_cast<const float4*>(H)[i4];
  v.x += rowv + pc[((t + 0) & 63) * 64 + c];
  v.y += rowv + pc[((t + 1) & 63) * 64 + c];
  v.z += rowv + pc[((t + 2) & 63) * 64 + c];
  v.w += rowv + pc[((t + 3) & 63) * 64 + c];
  reinterpret_cast<float4*>(H)[i4] = v;
}

__global__ __launch_bounds__(256) void ln_kernel(
    const float* __restrict__ X, const float* __restrict__ g,
    const float* __restrict__ be, float* __restrict__ Y)
{
  long i = (long)blockIdx.x * 256 + threadIdx.x;
  if (i >= (long)BATCH * TLEN) return;
  long b = i / TLEN;
  int t = (int)(i % TLEN);
  const float* xp = X + b * 64L * TLEN + t;
  float m = 0.f;
  for (int c = 0; c < 64; ++c) m += xp[(long)c * TLEN];
  m *= (1.f / 64.f);
  float v = 0.f;
  for (int c = 0; c < 64; ++c) { float d = xp[(long)c * TLEN] - m; v += d * d; }
  float inv = rsqrtf(v * (1.f / 64.f) + 1e-5f);
  float* yp = Y + b * 64L * TLEN + t;
  for (int c = 0; c < 64; ++c)
    yp[(long)c * TLEN] = (xp[(long)c * TLEN] - m) * inv * g[c] + be[c];
}

// ---- local attention (heads 0..1, windows of 64 with +-1 neighbors) ----
__global__ __launch_bounds__(64) void local_attn_kernel(
    const float* __restrict__ Q, const float* __restrict__ KV,
    float* __restrict__ AO)
{
  __shared__ float kl[192 * 8];
  __shared__ float vl[192 * 8];
  const int wwin = blockIdx.x & 127;
  const int bh   = blockIdx.x >> 7;
  const int h    = bh & 1;
  const int b    = bh >> 1;
  const int ch   = h * 8;
  const float* Kb = KV + (long)b * 128 * TLEN + (long)ch * TLEN;
  const float* Vb = KV + (long)b * 128 * TLEN + (long)(64 + ch) * TLEN;
  const int j0 = (wwin == 0) ? 64 : 0;
  const int j1 = (wwin == NBUCKET - 1) ? 128 : 192;
  const int tbase = (wwin - 1) * 64;
  // per-lane async global->LDS scatter copies (ASYNCcnt) when available
  for (int i = threadIdx.x; i < (j1 - j0) * 8; i += 64) {
    int j = j0 + (i >> 3);
    int d = i & 7;
    int t = tbase + j;
    async_copy_b32(&kl[j * 8 + d], &Kb[(long)d * TLEN + t]);
    async_copy_b32(&vl[j * 8 + d], &Vb[(long)d * TLEN + t]);
  }
  async_wait();
  __syncthreads();
  const int tq = wwin * 64 + threadIdx.x;
  const float* Qp = Q + (long)b * 64 * TLEN + (long)ch * TLEN + tq;
  float qr[8];
  #pragma unroll
  for (int d = 0; d < 8; ++d) qr[d] = Qp[(long)d * TLEN];
  float mx = -3.4e38f, ssum = 0.f;
  float o[8] = {0.f, 0.f, 0.f, 0.f, 0.f, 0.f, 0.f, 0.f};
  for (int j = j0; j < j1; ++j) {
    float r = 0.f;
    #pragma unroll
    for (int d = 0; d < 8; ++d) r += qr[d] * kl[j * 8 + d];
    r *= 0.35355339059327f;   // DH^-0.5
    float p;
    if (r > mx) {
      float sc = expf(mx - r);
      ssum *= sc;
      #pragma unroll
      for (int d = 0; d < 8; ++d) o[d] *= sc;
      mx = r; p = 1.f;
    } else p = expf(r - mx);
    ssum += p;
    #pragma unroll
    for (int d = 0; d < 8; ++d) o[d] += p * vl[j * 8 + d];
  }
  float inv = 1.f / ssum;
  float* Op = AO + (long)b * 64 * TLEN + (long)ch * TLEN + tq;
  #pragma unroll
  for (int d = 0; d < 8; ++d) Op[(long)d * TLEN] = o[d] * inv;
}

// ---- sinkhorn: per-bucket mean of q/k (heads 2..7) ----
__global__ __launch_bounds__(256) void bucket_mean_kernel(
    const float* __restrict__ Q, const float* __restrict__ KV,
    float* __restrict__ SQ, float* __restrict__ SK)
{
  int idx = blockIdx.x * 256 + threadIdx.x;
  if (idx >= BATCH * 6 * NBUCKET * 8) return;
  int d  = idx & 7;
  int nb = (idx >> 3) & 127;
  int bh = idx >> 10;               // b*6 + hs
  int hs = bh % 6;
  int b  = bh / 6;
  int ch = 16 + hs * 8 + d;
  const float* qp = Q  + (long)b * 64  * TLEN + (long)ch * TLEN + nb * 64;
  const float* kp = KV + (long)b * 128 * TLEN + (long)ch * TLEN + nb * 64;
  float s1 = 0.f, s2 = 0.f;
  for (int j = 0; j < 64; ++j) { s1 += qp[j]; s2 += kp[j]; }
  SQ[idx] = s1 * (1.f / 64.f);
  SK[idx] = s2 * (1.f / 64.f);
}

// ---- sinkhorn routing: softmax(R/temp) argmax + its probability ----
__global__ __launch_bounds__(128) void sink_sort_kernel(
    const float* __restrict__ SQ, const float* __restrict__ SK,
    int* __restrict__ IDX, float* __restrict__ VAL)
{
  __shared__ __attribute__((aligned(16))) float skl[NBUCKET * 8];
  const int bh = blockIdx.x;
  // contiguous 4KB block copy: 2 x b128 per thread (async when available)
  for (int i = threadIdx.x * 4; i < NBUCKET * 8; i += 128 * 4)
    async_copy_b128(&skl[i], &SK[(long)bh * NBUCKET * 8 + i]);
  async_wait();
  __syncthreads();
  const int i = threadIdx.x;
  float qr[8];
  #pragma unroll
  for (int d = 0; d < 8; ++d) qr[d] = SQ[(long)bh * NBUCKET * 8 + i * 8 + d];
  float mx = -3.4e38f; int am = 0;
  for (int j = 0; j < NBUCKET; ++j) {
    float r = 0.f;
    #pragma unroll
    for (int d = 0; d < 8; ++d) r += qr[d] * skl[j * 8 + d];
    r *= 0.125f;                      // ADIM^-0.5
    if (r > mx) { mx = r; am = j; }
  }
  float ssum = 0.f;
  for (int j = 0; j < NBUCKET; ++j) {
    float r = 0.f;
    #pragma unroll
    for (int d = 0; d < 8; ++d) r += qr[d] * skl[j * 8 + d];
    r *= 0.125f;
    ssum += expf((r - mx) * (1.f / 0.75f));   // / SORT_TEMP
  }
  IDX[bh * NBUCKET + i] = am;
  VAL[bh * NBUCKET + i] = 1.f / ssum;         // softmax value at argmax
}

// ---- sinkhorn attention: own bucket + gathered (scaled) bucket ----
__global__ __launch_bounds__(64) void sink_attn_kernel(
    const float* __restrict__ Q, const float* __restrict__ KV,
    const int* __restrict__ IDX, const float* __restrict__ VAL,
    float* __restrict__ AO)
{
  __shared__ float kl[128 * 8];
  __shared__ float vl[128 * 8];
  const int nb = blockIdx.x & 127;
  const int bh = blockIdx.x >> 7;       // b*6 + hs
  const int hs = bh % 6;
  const int b  = bh / 6;
  const int ch = 16 + hs * 8;
  const int g  = IDX[bh * NBUCKET + nb];
  const float val = VAL[bh * NBUCKET + nb];
  const float* Kb = KV + (long)b * 128 * TLEN + (long)ch * TLEN;
  const float* Vb = KV + (long)b * 128 * TLEN + (long)(64 + ch) * TLEN;
  for (int i = threadIdx.x; i < 64 * 8; i += 64) {
    int j = i >> 3, d = i & 7;
    kl[j * 8 + d]        = Kb[(long)d * TLEN + g * 64 + j] * val;   // gathered
    vl[j * 8 + d]        = Vb[(long)d * TLEN + g * 64 + j] * val;
    kl[(64 + j) * 8 + d] = Kb[(long)d * TLEN + nb * 64 + j];        // own
    vl[(64 + j) * 8 + d] = Vb[(long)d * TLEN + nb * 64 + j];
  }
  __syncthreads();
  const int tq = nb * 64 + threadIdx.x;
  const float* Qp = Q + (long)b * 64 * TLEN + (long)ch * TLEN + tq;
  float qr[8];
  #pragma unroll
  for (int d = 0; d < 8; ++d) qr[d] = Qp[(long)d * TLEN];
  float mx = -3.4e38f, ssum = 0.f;
  float o[8] = {0.f, 0.f, 0.f, 0.f, 0.f, 0.f, 0.f, 0.f};
  for (int j = 0; j < 128; ++j) {
    float r = 0.f;
    #pragma unroll
    for (int d = 0; d < 8; ++d) r += qr[d] * kl[j * 8 + d];
    r *= 0.35355339059327f;
    float p;
    if (r > mx) {
      float sc = expf(mx - r);
      ssum *= sc;
      #pragma unroll
      for (int d = 0; d < 8; ++d) o[d] *= sc;
      mx = r; p = 1.f;
    } else p = expf(r - mx);
    ssum += p;
    #pragma unroll
    for (int d = 0; d < 8; ++d) o[d] += p * vl[j * 8 + d];
  }
  float inv = 1.f / ssum;
  float* Op = AO + (long)b * 64 * TLEN + (long)ch * TLEN + tq;
  #pragma unroll
  for (int d = 0; d < 8; ++d) Op[(long)d * TLEN] = o[d] * inv;
}

// =====================================================================
// Host-side orchestration
// =====================================================================
static inline const float* PF(void* const* d_in, int i) {
  return (const float*)d_in[i];
}

static void launch_gemm(hipStream_t st, const float* Wm, const float* X,
                        const float* bias, float* Y, int M, int K, int N,
                        int ldx, int ldy, long sxb, long syb, int act, int accum) {
  dim3 grid((N + 63) / 64, (M + 63) / 64, BATCH);
  wmma_gemm_kernel<<<grid, dim3(256), 0, st>>>(Wm, X, bias, Y, M, K, N,
                                               ldx, ldy, sxb, syb, act, accum);
}

static void launch_dconv(hipStream_t st, const float* Wt, const float* X,
                         const float* bias, float* Y, int C, int taps,
                         int ar, int accum) {
  dim3 grid((SEQ + 63) / 64, (C + 63) / 64, BATCH);
  wmma_dconv_kernel<<<grid, dim3(256), 0, st>>>(Wt, X, bias, Y, C, SEQ, taps,
                                                ar, accum);
}

static void launch_bnrelu(hipStream_t st, const float* x, const float* g,
                          const float* be, const float* rm, const float* rv,
                          float* y, int L) {
  long total4 = (long)BATCH * L * (SEQ / 4);
  bnrelu_kernel<<<dim3((unsigned)((total4 + 255) / 256)), dim3(256), 0, st>>>(
      x, g, be, rm, rv, y, L);
}

static void launch_copy_crop(hipStream_t st, const float* src, float* feat,
                             int Lsrc, int nch, int chOff) {
  long total4 = (long)BATCH * nch * (TLEN / 4);
  copy_crop_kernel<<<dim3((unsigned)((total4 + 255) / 256)), dim3(256), 0, st>>>(
      src, feat, Lsrc, nch, chOff);
}

static void launch_ln(hipStream_t st, const float* X, const float* g,
                      const float* be, float* Y) {
  ln_kernel<<<dim3((BATCH * TLEN + 255) / 256), dim3(256), 0, st>>>(X, g, be, Y);
}

// jax tree-flatten (sorted dict keys) parameter index map:
// 0:x  1:conv1a_b 2:conv1a_w 3:conv1b_b 4:conv1b_w 5:conv2_b 6:conv2_w
// enc base e: e+0 conv1_b, e+1 conv1_w, e+2+2j convs[j].b, e+3+2j convs[j].w,
//   e+10+12i+{0..11}: bn1_b,bn1_g,bn1_rm,bn1_rv,bn2_b,bn2_g,bn2_rm,bn2_rv,
//                     conv1_b,conv1_w,conv2_b,conv2_w,  e+202 skip_b, e+203 skip_w
// enc1 base 7, enc2 base 211, layers base 415 (12 leaves/layer, sorted:
//   ff1_b,ff1_w,ff2_b,ff2_w,ln1_b,ln1_g,ln2_b,ln2_g,wkv,wo,wo_b,wq)
// 487 norm_b 488 norm_g 489 pos_col 490 pos_row 491 usage_b 492 usage_w
static void run_encoder(hipStream_t st, void* const* d_in, int base, int L,
                        const float* x, float* conv, float* t1, float* t2,
                        float* skip) {
  static const int Wtaps[16] = {11,11,11,11,11,11,11,11,21,21,21,21,21,21,21,21};
  static const int Ardil[16] = {1,1,1,1,4,4,4,4,10,10,10,10,20,20,20,20};
  const long LS = (long)L * SEQ;
  // conv = pconv(x, conv1);  skip = pconv(conv, skip)
  launch_gemm(st, PF(d_in, base + 1), x, PF(d_in, base + 0), conv,
              L, 4, SEQ, SEQ, SEQ, 4L * SEQ, LS, ACT_NONE, 0);
  launch_gemm(st, PF(d_in, base + 203), conv, PF(d_in, base + 202), skip,
              L, L, SEQ, SEQ, SEQ, LS, LS, ACT_NONE, 0);
  int j = 0;
  for (int i = 0; i < 16; ++i) {
    int rb = base + 10 + 12 * i;
    launch_bnrelu(st, conv, PF(d_in, rb + 1), PF(d_in, rb + 0),
                  PF(d_in, rb + 2), PF(d_in, rb + 3), t1, L);
    launch_dconv(st, PF(d_in, rb + 9), t1, PF(d_in, rb + 8), t2,
                 L, Wtaps[i], Ardil[i], 0);
    launch_bnrelu(st, t2, PF(d_in, rb + 5), PF(d_in, rb + 4),
                  PF(d_in, rb + 6), PF(d_in, rb + 7), t1, L);
    launch_dconv(st, PF(d_in, rb + 11), t1, PF(d_in, rb + 10), conv,
                 L, Wtaps[i], Ardil[i], 1);            // conv += h
    if (((i + 1) & 3) == 0) {                          // skip += pconv(conv)
      launch_gemm(st, PF(d_in, base + 3 + 2 * j), conv,
                  PF(d_in, base + 2 + 2 * j), skip,
                  L, L, SEQ, SEQ, SEQ, LS, LS, ACT_NONE, 1);
      ++j;
    }
  }
}

extern "C" void kernel_launch(void* const* d_in, const int* in_sizes, int n_in,
                              void* d_out, int out_size, void* d_ws, size_t ws_size,
                              hipStream_t stream) {
  (void)in_sizes; (void)n_in; (void)out_size; (void)ws_size;
  const float* x = (const float*)d_in[0];
  float* w = (float*)d_ws;
  long off = 0;
  auto alloc = [&](long n) { float* p = w + off; off += n; return p; };
  const long SBIG = (long)BATCH * 128 * SEQ;
  float* e_conv = alloc(SBIG);
  float* e_t1   = alloc(SBIG);
  float* e_t2   = alloc(SBIG);
  float* e_skip = alloc(SBIG);
  float* feat = alloc((long)BATCH * FEAT_CH * TLEN);
  float* H    = alloc((long)BATCH * 64 * TLEN);
  float* HN   = alloc((long)BATCH * 64 * TLEN);
  float* Qb   = alloc((long)BATCH * 64 * TLEN);
  float* KV   = alloc((long)BATCH * 128 * TLEN);
  float* AO   = alloc((long)BATCH * 64 * TLEN);
  float* FF   = alloc((long)BATCH * 256 * TLEN);
  float* SQ   = alloc((long)BATCH * 6 * NBUCKET * 8);
  float* SK   = alloc((long)BATCH * 6 * NBUCKET * 8);
  float* VAL  = alloc((long)BATCH * 6 * NBUCKET);
  int*   IDXp = (int*)alloc((long)BATCH * 6 * NBUCKET);

  // ---- encoders -> feat (cropped to TLEN) ----
  run_encoder(stream, d_in, 7, 128, x, e_conv, e_t1, e_t2, e_skip);
  launch_copy_crop(stream, e_skip, feat, 128, 128, 0);
  run_encoder(stream, d_in, 211, 64, x, e_conv, e_t1, e_t2, e_skip);
  launch_copy_crop(stream, e_skip, feat, 64, 64, 128);
  // feat2 = conv1b(conv1a(x))
  launch_gemm(stream, PF(d_in, 2), x, PF(d_in, 1), e_conv,
              32, 4, SEQ, SEQ, SEQ, 4L * SEQ, 32L * SEQ, ACT_NONE, 0);
  launch_gemm(stream, PF(d_in, 4), e_conv, PF(d_in, 3), e_t1,
              32, 32, SEQ, SEQ, SEQ, 32L * SEQ, 32L * SEQ, ACT_NONE, 0);
  launch_copy_crop(stream, e_t1, feat, 32, 32, 192);

  // ---- emb = conv2(feat); h = emb + pos ----
  launch_gemm(stream, PF(d_in, 6), feat, PF(d_in, 5), H,
              64, FEAT_CH, TLEN, TLEN, TLEN,
              (long)FEAT_CH * TLEN, 64L * TLEN, ACT_NONE, 0);
  add_pe_kernel<<<dim3((unsigned)(((long)BATCH * 64 * (TLEN / 4) + 255) / 256)),
                  dim3(256), 0, stream>>>(H, PF(d_in, 490), PF(d_in, 489));

  // ---- 6 transformer layers ----
  for (int l = 0; l < 6; ++l) {
    int lb = 415 + 12 * l;
    launch_ln(stream, H, PF(d_in, lb + 5), PF(d_in, lb + 4), HN);   // ln1
    launch_gemm(stream, PF(d_in, lb + 11), HN, nullptr, Qb,          // wq
                64, 64, TLEN, TLEN, TLEN, 64L * TLEN, 64L * TLEN, ACT_NONE, 0);
    launch_gemm(stream, PF(d_in, lb + 8), HN, nullptr, KV,           // wkv
                128, 64, TLEN, TLEN, TLEN, 64L * TLEN, 128L * TLEN, ACT_NONE, 0);
    local_attn_kernel<<<dim3(BATCH * 2 * NBUCKET), dim3(64), 0, stream>>>(Qb, KV, AO);
    bucket_mean_kernel<<<dim3((BATCH * 6 * NBUCKET * 8 + 255) / 256),
                         dim3(256), 0, stream>>>(Qb, KV, SQ, SK);
    sink_sort_kernel<<<dim3(BATCH * 6), dim3(128), 0, stream>>>(SQ, SK, IDXp, VAL);
    sink_attn_kernel<<<dim3(BATCH * 6 * NBUCKET), dim3(64), 0, stream>>>(
        Qb, KV, IDXp, VAL, AO);
    launch_gemm(stream, PF(d_in, lb + 9), AO, PF(d_in, lb + 10), H,  // h += wo
                64, 64, TLEN, TLEN, TLEN, 64L * TLEN, 64L * TLEN, ACT_NONE, 1);
    launch_ln(stream, H, PF(d_in, lb + 7), PF(d_in, lb + 6), HN);    // ln2
    launch_gemm(stream, PF(d_in, lb + 1), HN, PF(d_in, lb + 0), FF,  // ff1+gelu
                256, 64, TLEN, TLEN, TLEN, 64L * TLEN, 256L * TLEN, ACT_GELU, 0);
    launch_gemm(stream, PF(d_in, lb + 3), FF, PF(d_in, lb + 2), H,   // h += ff2
                64, 256, TLEN, TLEN, TLEN, 256L * TLEN, 64L * TLEN, ACT_NONE, 1);
  }

  // ---- final norm + usage projection, fused output crop (cols 3596..4596) ----
  launch_ln(stream, H, PF(d_in, 488), PF(d_in, 487), HN);
  launch_gemm(stream, PF(d_in, 492), HN + 3596, PF(d_in, 491), (float*)d_out,
              15, 64, 1000, TLEN, 1000, 64L * TLEN, 15L * 1000, ACT_NONE, 0);
}
